// LiquidODECore_24927990186047
// MI455X (gfx1250) — matrix operand order repaired and checked by reading, not compiled, not run on hardware
//
#include <hip/hip_runtime.h>
#include <math.h>

#define HD 64
#define NPTS 110592          // 48^3 points per batch
#define ROWS_TOTAL 221184    // B * NPTS
#define WSTR 72              // f16 elements per weight row (k-major), padded
#define HSTR 72              // f16 elements per h row (m-major), padded
#define H32S 66              // f32 stride for readout staging

#define L2E  1.44269504088896340736f   // log2(e)
#define LN2  0.69314718055994530942f   // ln(2)
#define TAU_SCALE (9.9f * LN2)         // (MAX_TAU-MIN_TAU) * ln2, folds log2->ln

typedef __attribute__((ext_vector_type(16))) _Float16 v16h;
typedef __attribute__((ext_vector_type(8)))  _Float16 v8h;
typedef __attribute__((ext_vector_type(8)))  float    v8f;

union ABfrag { v16h v; v8h h[2]; };

// Compiler-only ordering fence: LDS ops from one wave are executed in order by
// the DS unit (CDNA5 ISA 7.3), so store->load visibility within a wave needs
// no hardware wait -- only prevent the compiler from reordering/hoisting.
#define LDS_COMPILER_FENCE() asm volatile("" ::: "memory")

__global__ __launch_bounds__(128)
void liquid_ode_kernel(const float* __restrict__ coords,
                       const float* __restrict__ Wh,
                       const float* __restrict__ Uh,
                       const float* __restrict__ bh,
                       const float* __restrict__ Wtau,
                       const float* __restrict__ Utau,
                       const float* __restrict__ btau,
                       const float* __restrict__ Wout,
                       const float* __restrict__ bout,
                       float* __restrict__ out)
{
    // LDS: weights (k-major, f16) shared by all 4 waves; per-wave h staging.
    __shared__ __align__(16) _Float16 sWh[HD * WSTR];
    __shared__ __align__(16) _Float16 sWt[HD * WSTR];
    __shared__ __align__(16) float    sWout[3 * HD];
    __shared__ __align__(16) float    sCoord[4][48];
    __shared__ __align__(16) _Float16 sH16[4][16 * HSTR];
    __shared__ __align__(16) float    sH32[4][16 * H32S];

    const int tid  = threadIdx.x;
    const int w    = tid >> 5;      // wave id in block (0..3)
    const int lane = tid & 31;
    const int lo   = lane & 15;
    const int hi   = lane >> 4;

    // ---- Stage weights: sW[n*WSTR + k] = W[n][k] as f16 (B-matrix layout) ----
    for (int idx = tid; idx < HD * HD; idx += 128) {
        const int n = idx >> 6, k = idx & 63;
        sWh[n * WSTR + k] = (_Float16)Wh[idx];
        sWt[n * WSTR + k] = (_Float16)Wtau[idx];
    }
    for (int idx = tid; idx < 3 * HD; idx += 128) sWout[idx] = Wout[idx];

    // ---- Stage this wave's 16 coords ----
    const long gbase = (long)blockIdx.x * 64 + (long)w * 16;
    if (lane < 16) {
        const float* cp = coords + (gbase + lane) * 3;
        sCoord[w][lane * 3 + 0] = cp[0];
        sCoord[w][lane * 3 + 1] = cp[1];
        sCoord[w][lane * 3 + 2] = cp[2];
    }
    __syncthreads();

    // ---- Precompute input projections u_h, u_tau in C-fragment layout ----
    // fragment element (j, v, lane): row m = hi*8+v, col c = j*16+lo
    float uh[4][8], ut[4][8];
#pragma unroll
    for (int j = 0; j < 4; ++j) {
        const int c = j * 16 + lo;
        const float Uh0 = Uh[c * 3 + 0], Uh1 = Uh[c * 3 + 1], Uh2 = Uh[c * 3 + 2];
        const float Ut0 = Utau[c * 3 + 0], Ut1 = Utau[c * 3 + 1], Ut2 = Utau[c * 3 + 2];
        const float bhc = bh[c], btc = btau[c];
#pragma unroll
        for (int v = 0; v < 8; ++v) {
            const int m = hi * 8 + v;
            const float x = sCoord[w][m * 3 + 0];
            const float y = sCoord[w][m * 3 + 1];
            const float z = sCoord[w][m * 3 + 2];
            uh[j][v] = bhc + x * Uh0 + y * Uh1 + z * Uh2;
            ut[j][v] = btc + x * Ut0 + y * Ut1 + z * Ut2;
        }
    }

    // Per-lane invariant LDS offsets.
    // B 32x16 f16 layout: lane holds column N = lo, K split by lane half (like A).
    const _Float16* wh0 = &sWh[lo * WSTR + hi * 8];         // kc = 0
    const _Float16* wh1 = &sWh[lo * WSTR + 32 + hi * 8];    // kc = 1
    const _Float16* wt0 = &sWt[lo * WSTR + hi * 8];
    const _Float16* wt1 = &sWt[lo * WSTR + 32 + hi * 8];
    const _Float16* ap  = &sH16[w][lo * HSTR + hi * 8];
    _Float16*       hsp = &sH16[w][hi * 8 * HSTR + lo];

    // ---- Hidden state: 4 C-fragments of 16x16 f32, start at zero ----
    float hreg[4][8];
#pragma unroll
    for (int j = 0; j < 4; ++j)
#pragma unroll
        for (int v = 0; v < 8; ++v) hreg[j][v] = 0.0f;

    // ---- 8 Euler steps of the liquid ODE ----
    for (int t = 0; t < 8; ++t) {
        // dump h to LDS as f16 (row-major [m][k], stride HSTR)
#pragma unroll
        for (int j = 0; j < 4; ++j)
#pragma unroll
            for (int v = 0; v < 8; ++v)
                hsp[v * HSTR + j * 16] = (_Float16)hreg[j][v];

        LDS_COMPILER_FENCE();   // in-order DS per wave handles the hardware side

        // reload as A fragments (16x32 f16): lane row m=lo, K split by lane half
        ABfrag A0, A1;
        A0.h[0] = *(const v8h*)(ap);        // K =      hi*8 .. +7
        A0.h[1] = *(const v8h*)(ap + 16);   // K = 16 + hi*8 .. +7
        A1.h[0] = *(const v8h*)(ap + 32);   // K = 32 + hi*8 .. +7
        A1.h[1] = *(const v8h*)(ap + 48);   // K = 48 + hi*8 .. +7

#pragma unroll
        for (int j = 0; j < 4; ++j) {
            // B fragments for this N-tile, reloaded from LDS (keeps VGPRs < 256)
            ABfrag Bh0, Bh1, Bt0, Bt1;
            Bh0.h[0] = *(const v8h*)(wh0 + j * 16 * WSTR);
            Bh0.h[1] = *(const v8h*)(wh0 + j * 16 * WSTR + 16);
            Bh1.h[0] = *(const v8h*)(wh1 + j * 16 * WSTR);
            Bh1.h[1] = *(const v8h*)(wh1 + j * 16 * WSTR + 16);
            Bt0.h[0] = *(const v8h*)(wt0 + j * 16 * WSTR);
            Bt0.h[1] = *(const v8h*)(wt0 + j * 16 * WSTR + 16);
            Bt1.h[0] = *(const v8h*)(wt1 + j * 16 * WSTR);
            Bt1.h[1] = *(const v8h*)(wt1 + j * 16 * WSTR + 16);

            v8f ah, at;
#pragma unroll
            for (int v = 0; v < 8; ++v) { ah[v] = uh[j][v]; at[v] = ut[j][v]; }

            ah = __builtin_amdgcn_wmma_f32_16x16x32_f16(false, A0.v, false, Bh0.v,
                                                        (short)0, ah, false, false);
            ah = __builtin_amdgcn_wmma_f32_16x16x32_f16(false, A1.v, false, Bh1.v,
                                                        (short)0, ah, false, false);
            at = __builtin_amdgcn_wmma_f32_16x16x32_f16(false, A0.v, false, Bt0.v,
                                                        (short)0, at, false, false);
            at = __builtin_amdgcn_wmma_f32_16x16x32_f16(false, A1.v, false, Bt1.v,
                                                        (short)0, at, false, false);

            // elementwise liquid update -- raw v_exp/v_log/v_rcp paths.
            // Inputs are clipped to +/-50, so 1+exp(.) is never denormal and
            // never overflows: the accurate-libm fixup code is unnecessary.
#pragma unroll
            for (int v = 0; v < 8; ++v) {
                float x = at[v];
                x = fminf(fmaxf(x, -50.0f), 50.0f);                 // v_med3
                const float e  = __builtin_amdgcn_exp2f(x * L2E);   // exp(x)
                // tau = 0.1 + softplus(x)*9.9 = 0.1 + log2(1+e)*(9.9*ln2)
                float tau = 0.1f + __builtin_amdgcn_logf(1.0f + e) * TAU_SCALE;
                tau = fmaxf(tau, 1e-6f);
                // sigmoid(ah) = 1 / (1 + exp(-ah))
                const float en = __builtin_amdgcn_exp2f(-ah[v] * L2E);
                const float fh = __builtin_amdgcn_rcpf(1.0f + en);
                const float hv = hreg[j][v];
                hreg[j][v] = hv + 0.125f * (fh - hv * __builtin_amdgcn_rcpf(tau));
            }
        }
    }

    // ---- Readout: v = tanh(h @ Wout^T + bout) * 10 ----
#pragma unroll
    for (int j = 0; j < 4; ++j)
#pragma unroll
        for (int v = 0; v < 8; ++v)
            sH32[w][(hi * 8 + v) * H32S + j * 16 + lo] = hreg[j][v];

    LDS_COMPILER_FENCE();

    {
        const int mrow = lo;
        float s0 = bout[0], s1 = bout[1], s2 = bout[2];
#pragma unroll 4
        for (int k = 0; k < HD; ++k) {
            const float hk = sH32[w][mrow * H32S + k];
            s0 += hk * sWout[0 * HD + k];
            s1 += hk * sWout[1 * HD + k];
            s2 += hk * sWout[2 * HD + k];
        }
        if (lane < 16) {
            const long g = gbase + mrow;
            const int  b = (int)(g / NPTS);
            const int  n = (int)(g % NPTS);
            // tanh(s)*10 via exp2: tanh(s) = 1 - 2/(exp(2s)+1)
            const float e0 = __builtin_amdgcn_exp2f(s0 * (2.0f * L2E));
            const float e1 = __builtin_amdgcn_exp2f(s1 * (2.0f * L2E));
            const float e2 = __builtin_amdgcn_exp2f(s2 * (2.0f * L2E));
            const float t0 = 1.0f - 2.0f * __builtin_amdgcn_rcpf(e0 + 1.0f);
            const float t1 = 1.0f - 2.0f * __builtin_amdgcn_rcpf(e1 + 1.0f);
            const float t2 = 1.0f - 2.0f * __builtin_amdgcn_rcpf(e2 + 1.0f);
            out[((long)(b * 3 + 0)) * NPTS + n] = t0 * 10.0f;
            out[((long)(b * 3 + 1)) * NPTS + n] = t1 * 10.0f;
            out[((long)(b * 3 + 2)) * NPTS + n] = t2 * 10.0f;
        }
    }
}

extern "C" void kernel_launch(void* const* d_in, const int* in_sizes, int n_in,
                              void* d_out, int out_size, void* d_ws, size_t ws_size,
                              hipStream_t stream) {
    const float* coords = (const float*)d_in[0];
    const float* Wh     = (const float*)d_in[1];
    const float* Uh     = (const float*)d_in[2];
    const float* bh     = (const float*)d_in[3];
    const float* Wtau   = (const float*)d_in[4];
    const float* Utau   = (const float*)d_in[5];
    const float* btau   = (const float*)d_in[6];
    const float* Wout   = (const float*)d_in[7];
    const float* bout   = (const float*)d_in[8];

    dim3 grid(ROWS_TOTAL / 64);   // 64 points per block (4 waves x 16)
    dim3 block(128);
    hipLaunchKernelGGL(liquid_ode_kernel, grid, block, 0, stream,
                       coords, Wh, Uh, bh, Wtau, Utau, btau, Wout, bout,
                       (float*)d_out);
}
